// MOEFFN_27427661152612
// MI455X (gfx1250) — matrix operand order repaired and checked by reading, not compiled, not run on hardware
//
#include <hip/hip_runtime.h>

// ---------------- problem constants (from reference) ----------------
#define T_TOK 4096            // B*S
#define D_DIM 1024
#define H_DIM 2816
#define E_NUM 8
#define TM    32              // tokens per workgroup tile (2 subtiles of 16)
#define HT    256             // H chunk per iteration
#define NITER (H_DIM / HT)    // 11
#define LDX   (D_DIM + 8)     // ushort stride of X tile in LDS (bank-spread pad)
#define LDH   (HT + 8)        // ushort stride of h tile in LDS
#define BLK   512             // ushorts per swizzled 16x32 weight block

typedef __attribute__((ext_vector_type(16))) __bf16 v16bf;
typedef __attribute__((ext_vector_type(8)))  float  v8f;

union Frag { v16bf v; uint4 q[2]; };

__device__ __forceinline__ unsigned short f2bf(float f) {
  unsigned int u = __float_as_uint(f);
  u += 0x7FFFu + ((u >> 16) & 1u);     // round-to-nearest-even
  return (unsigned short)(u >> 16);
}

// ---------------- kernel 1: zero output + expert counts ----------------
__global__ void zero_kernel(float* __restrict__ out, int n, int* __restrict__ counts) {
  int i = blockIdx.x * blockDim.x + threadIdx.x;
  if (i < E_NUM) counts[i] = 0;
  for (int j = i; j < n; j += gridDim.x * blockDim.x) out[j] = 0.f;
}

// ---------------- kernel 2a: fp32 -> bf16 linear (for x) ----------------
__global__ void cvt_kernel(const float* __restrict__ src, unsigned short* __restrict__ dst, int n4) {
  int i = blockIdx.x * blockDim.x + threadIdx.x;
  const float4* s4 = (const float4*)src;
  uint2* d2 = (uint2*)dst;
  for (int j = i; j < n4; j += gridDim.x * blockDim.x) {
    float4 v = s4[j];
    uint2 o;
    o.x = (unsigned)f2bf(v.x) | ((unsigned)f2bf(v.y) << 16);
    o.y = (unsigned)f2bf(v.z) | ((unsigned)f2bf(v.w) << 16);
    d2[j] = o;
  }
}

// ---------------- kernel 2b: fp32 -> bf16, swizzled into WMMA B-fragment blocks ----
// src viewed as [Rtot][C] row-major (Rtot % 16 == 0, C % 32 == 0).
// Block b = rowTile * (C/32) + kblk holds a 16x32 tile as 512 bf16:
//   dst[b*512 + L*16 + j] = src[rowTile*16 + (L&15)][kblk*32 + (L>>4)*8 + j + (j&8)]
__global__ void cvt_swz_kernel(const float* __restrict__ src, unsigned short* __restrict__ dst,
                               int Rtot, int C) {
  const int kb = C >> 5;
  const int nChunk = (Rtot >> 4) * kb * 32;  // 16-elem chunks
  int i0 = blockIdx.x * blockDim.x + threadIdx.x;
  for (int i = i0; i < nChunk; i += gridDim.x * blockDim.x) {
    const int L = i & 31;
    const int b = i >> 5;
    const int kblk = b % kb;
    const int rowTile = b / kb;
    const float* s = src + (size_t)(rowTile * 16 + (L & 15)) * C + kblk * 32 + (L >> 4) * 8;
    unsigned short tmp[16];
#pragma unroll
    for (int j = 0; j < 16; j += 4) {
      float4 v = *(const float4*)(s + j + (j & 8));
      tmp[j + 0] = f2bf(v.x); tmp[j + 1] = f2bf(v.y);
      tmp[j + 2] = f2bf(v.z); tmp[j + 3] = f2bf(v.w);
    }
    uint4* d = (uint4*)(dst + (size_t)i * 16);
    d[0] = ((const uint4*)tmp)[0];
    d[1] = ((const uint4*)tmp)[1];
  }
}

// ---------------- kernel 3: gating + top-2 routing (one wave32 per token) ----------------
__global__ void gate_kernel(const float* __restrict__ x, const float* __restrict__ Wg,
                            int* __restrict__ counts, int* __restrict__ ids,
                            float* __restrict__ wts) {
  const int lane = threadIdx.x & 31;
  const int t = blockIdx.x * (blockDim.x >> 5) + (threadIdx.x >> 5);
  float acc[E_NUM];
#pragma unroll
  for (int e = 0; e < E_NUM; ++e) acc[e] = 0.f;
  const float* xr = x + (size_t)t * D_DIM;
  for (int i = 0; i < D_DIM / 32; ++i) {
    int d = i * 32 + lane;
    float xv = xr[d];
#pragma unroll
    for (int e = 0; e < E_NUM; ++e) acc[e] += xv * Wg[e * D_DIM + d];
  }
#pragma unroll
  for (int e = 0; e < E_NUM; ++e) {
#pragma unroll
    for (int off = 16; off > 0; off >>= 1) acc[e] += __shfl_xor(acc[e], off, 32);
  }
  if (lane == 0) {
    int e0 = 0; float s0 = acc[0];
#pragma unroll
    for (int e = 1; e < E_NUM; ++e) if (acc[e] > s0) { s0 = acc[e]; e0 = e; }
    int e1 = -1; float s1 = 0.f;
#pragma unroll
    for (int e = 0; e < E_NUM; ++e)
      if (e != e0 && (e1 < 0 || acc[e] > s1)) { s1 = acc[e]; e1 = e; }
    int slot0 = atomicAdd(&counts[e0], 1);
    ids[e0 * T_TOK + slot0] = t; wts[e0 * T_TOK + slot0] = s0;
    int slot1 = atomicAdd(&counts[e1], 1);
    ids[e1 * T_TOK + slot1] = t; wts[e1 * T_TOK + slot1] = s1;
  }
}

// ---------------- kernel 4: routed SwiGLU expert FFN, bf16 WMMA ----------------
// grid = (T_TOK/TM tiles, E_NUM experts), block = 512 (16 waves).
// Every B-fragment (weight block) is reused by both 16-token subtiles.
__global__ __launch_bounds__(512) void ffn_kernel(
    const unsigned short* __restrict__ xb,
    const unsigned short* __restrict__ w1b,
    const unsigned short* __restrict__ w2b,
    const unsigned short* __restrict__ w3b,
    const int* __restrict__ ids_g,
    const float* __restrict__ wts_g,
    const int* __restrict__ counts,
    float* __restrict__ out)
{
  const int e  = blockIdx.y;
  const int ne = counts[e];
  const int t0 = blockIdx.x * TM;
  if (t0 >= ne) return;

  extern __shared__ char smem[];
  unsigned short* Xs  = (unsigned short*)smem;                    // TM x LDX bf16
  unsigned short* Hs0 = Xs + (size_t)TM * LDX;                    // 2 x (TM x LDH) bf16
  int*   Ids = (int*)(Hs0 + (size_t)2 * TM * LDH);
  float* Wts = (float*)(Ids + TM);

  const int tid  = threadIdx.x;
  const int lane = tid & 31;
  const int w    = tid >> 5;   // wave 0..15

  if (tid < TM) {
    int idx = t0 + tid;
    int id = 0; float wt = 0.f;
    if (idx < ne) { id = ids_g[e * T_TOK + idx]; wt = wts_g[e * T_TOK + idx]; }
    Ids[tid] = id; Wts[tid] = wt;
  }
  __syncthreads();

  { // stage gathered X tile (bf16) into LDS: 32 rows x 16 segments of 64 cols
    const int row = tid >> 4;
    const int seg = tid & 15;
    const uint4* src = (const uint4*)(xb + (size_t)Ids[row] * D_DIM + seg * 64);
    uint4* dst = (uint4*)(Xs + row * LDX + seg * 64);
#pragma unroll
    for (int j = 0; j < 8; ++j) dst[j] = src[j];
  }
  __syncthreads();

  const int ln15 = lane & 15;
  const int lhi  = lane >> 4;

  v8f acc[2][4];
#pragma unroll
  for (int t = 0; t < 2; ++t)
#pragma unroll
    for (int s = 0; s < 4; ++s) acc[t][s] = v8f{0.f,0.f,0.f,0.f,0.f,0.f,0.f,0.f};

  // A fragment bases in LDS (16-bit A layout: K kb+0..7 / kb+16..23, kb = lhi*8)
  const unsigned short* paX0 = Xs + (ln15)      * LDX + lhi * 8;  // tokens 0-15
  const unsigned short* paX1 = Xs + (16 + ln15) * LDX + lhi * 8;  // tokens 16-31

  // Swizzled weight bases: row-tile stride = (C/32) blocks * 512 ushorts
  const size_t rtStrideW12 = (size_t)(D_DIM / 32) * BLK;
  const size_t rtStrideW3  = (size_t)(H_DIM / 32) * BLK;
  const unsigned short* pw1_base =
      w1b + (size_t)((e * H_DIM) >> 4) * rtStrideW12 + (size_t)w * rtStrideW12 + lane * 16;
  const unsigned short* pw2_base =
      w2b + (size_t)((e * H_DIM) >> 4) * rtStrideW12 + (size_t)w * rtStrideW12 + lane * 16;
  const unsigned short* pw3_base =
      w3b + ((size_t)((e * D_DIM) >> 4) + (size_t)w * 4) * rtStrideW3 + lane * 16;

  for (int it = 0; it < NITER; ++it) {
    const int hb = it * HT;
    unsigned short* Hw = Hs0 + (size_t)(it & 1) * TM * LDH;  // double-buffered h tile

    // ---- phase 1: gate/up = X*W1^T / X*W2^T for both token subtiles; B reused x2 ----
    v8f cg0 = v8f{0.f,0.f,0.f,0.f,0.f,0.f,0.f,0.f};
    v8f cg1 = cg0, cu0 = cg0, cu1 = cg0;
    const unsigned short* pw1 = pw1_base + (size_t)(hb >> 4) * rtStrideW12;
    const unsigned short* pw2 = pw2_base + (size_t)(hb >> 4) * rtStrideW12;
#pragma unroll 2
    for (int k0 = 0; k0 < D_DIM; k0 += 32) {
      Frag a0, a1, b1, b2;
      a0.q[0] = *(const uint4*)(paX0 + k0);
      a0.q[1] = *(const uint4*)(paX0 + k0 + 16);
      a1.q[0] = *(const uint4*)(paX1 + k0);
      a1.q[1] = *(const uint4*)(paX1 + k0 + 16);
      b1.q[0] = *(const uint4*)(pw1 + k0 * 16);
      b1.q[1] = *(const uint4*)(pw1 + k0 * 16 + 8);
      b2.q[0] = *(const uint4*)(pw2 + k0 * 16);
      b2.q[1] = *(const uint4*)(pw2 + k0 * 16 + 8);
      cg0 = __builtin_amdgcn_wmma_f32_16x16x32_bf16(false, a0.v, false, b1.v, (short)0, cg0, false, false);
      cg1 = __builtin_amdgcn_wmma_f32_16x16x32_bf16(false, a1.v, false, b1.v, (short)0, cg1, false, false);
      cu0 = __builtin_amdgcn_wmma_f32_16x16x32_bf16(false, a0.v, false, b2.v, (short)0, cu0, false, false);
      cu1 = __builtin_amdgcn_wmma_f32_16x16x32_bf16(false, a1.v, false, b2.v, (short)0, cu1, false, false);
    }
    // h = silu(gate) * up -> bf16 LDS  (fast sigmoid via v_rcp_f32)
#pragma unroll
    for (int v = 0; v < 8; ++v) {
      int m = v + 8 * lhi;
      float g0 = cg0[v];
      float h0 = g0 * __builtin_amdgcn_rcpf(1.f + __expf(-g0)) * cu0[v];
      Hw[(m)      * LDH + w * 16 + ln15] = f2bf(h0);
      float g1 = cg1[v];
      float h1 = g1 * __builtin_amdgcn_rcpf(1.f + __expf(-g1)) * cu1[v];
      Hw[(16 + m) * LDH + w * 16 + ln15] = f2bf(h1);
    }
    __syncthreads();   // only barrier per iteration (h buffer is double-buffered)

    // ---- phase 2: y[2x16, 64-d-chunk] += h[2x16, HT] * W3[d, h]^T; B reused x2 ----
    const unsigned short* paH0 = Hw + (ln15)      * LDH + lhi * 8;
    const unsigned short* paH1 = Hw + (16 + ln15) * LDH + lhi * 8;
#pragma unroll 2
    for (int ks = 0; ks < HT; ks += 32) {
      Frag a0, a1;
      a0.q[0] = *(const uint4*)(paH0 + ks);
      a0.q[1] = *(const uint4*)(paH0 + ks + 16);
      a1.q[0] = *(const uint4*)(paH1 + ks);
      a1.q[1] = *(const uint4*)(paH1 + ks + 16);
      const size_t kofs = (size_t)(hb + ks) * 16;
#pragma unroll
      for (int s = 0; s < 4; ++s) {
        Frag b;
        const unsigned short* pb = pw3_base + (size_t)s * rtStrideW3 + kofs;
        b.q[0] = *(const uint4*)pb;
        b.q[1] = *(const uint4*)(pb + 8);
        acc[0][s] = __builtin_amdgcn_wmma_f32_16x16x32_bf16(false, a0.v, false, b.v, (short)0, acc[0][s], false, false);
        acc[1][s] = __builtin_amdgcn_wmma_f32_16x16x32_bf16(false, a1.v, false, b.v, (short)0, acc[1][s], false, false);
      }
    }
  }

  // ---- epilogue: out[t] += score * y[t], fp32 global atomics ----
#pragma unroll
  for (int t01 = 0; t01 < 2; ++t01) {
#pragma unroll
    for (int s = 0; s < 4; ++s) {
#pragma unroll
      for (int v = 0; v < 8; ++v) {
        int m = v + 8 * lhi;
        int r = t01 * 16 + m;
        int t = Ids[r];
        float wt = Wts[r];
        int d = w * 64 + s * 16 + ln15;
        unsafeAtomicAdd(out + (size_t)t * D_DIM + d, wt * acc[t01][s][v]);
      }
    }
  }
}

// ---------------- host launcher ----------------
extern "C" void kernel_launch(void* const* d_in, const int* in_sizes, int n_in,
                              void* d_out, int out_size, void* d_ws, size_t ws_size,
                              hipStream_t stream) {
  const float* x  = (const float*)d_in[0];
  const float* Wg = (const float*)d_in[1];
  const float* W1 = (const float*)d_in[2];
  const float* W2 = (const float*)d_in[3];
  const float* W3 = (const float*)d_in[4];
  float* out = (float*)d_out;

  const size_t nW = (size_t)E_NUM * H_DIM * D_DIM;   // per weight tensor
  const size_t nX = (size_t)T_TOK * D_DIM;

  unsigned short* w1b = (unsigned short*)d_ws;
  unsigned short* w2b = w1b + nW;
  unsigned short* w3b = w2b + nW;
  unsigned short* xb  = w3b + nW;
  int*   ids    = (int*)(xb + nX);
  float* wts    = (float*)(ids + (size_t)E_NUM * T_TOK);
  int*   counts = (int*)(wts + (size_t)E_NUM * T_TOK);

  zero_kernel<<<2048, 256, 0, stream>>>(out, (int)nX, counts);
  // weights -> bf16 in WMMA B-fragment block order
  cvt_swz_kernel<<<4096, 256, 0, stream>>>(W1, w1b, E_NUM * H_DIM, D_DIM);
  cvt_swz_kernel<<<4096, 256, 0, stream>>>(W2, w2b, E_NUM * H_DIM, D_DIM);
  cvt_swz_kernel<<<4096, 256, 0, stream>>>(W3, w3b, E_NUM * D_DIM, H_DIM);
  // activations -> bf16 linear
  cvt_kernel<<<1024, 256, 0, stream>>>(x, xb, (int)(nX / 4));
  gate_kernel<<<T_TOK / 8, 256, 0, stream>>>(x, Wg, counts, ids, wts);

  const int smem = TM * LDX * 2 + 2 * TM * LDH * 2 + TM * 8;  // ~100 KB dynamic LDS
  ffn_kernel<<<dim3(T_TOK / TM, E_NUM), 512, smem, stream>>>(
      xb, w1b, w2b, w3b, ids, wts, counts, out);
}
